// MultiHeadAttention_17781164606186
// MI455X (gfx1250) — matrix operand order repaired and checked by reading, not compiled, not run on hardware
//
#include <hip/hip_runtime.h>
#include <hip/hip_bf16.h>

typedef __bf16 bf16_t;
typedef __attribute__((ext_vector_type(16))) __bf16 v16bf;
typedef __attribute__((ext_vector_type(8)))  __bf16 v8bf;
typedef __attribute__((ext_vector_type(8)))  float   v8f;

#define B_  8
#define C_  768
#define T_  1024
#define H_  12
#define KC_ 64

// ---------------- WMMA fragment helpers (wave32, gfx1250) ----------------
// A-matrix 16x32 bf16: lane holds row M = lane&15; K chunks of 8 at
// k0=(lane>>4)*8 and k0+16  (ISA 7.12.2, 16-bit A layout).
__device__ __forceinline__ v16bf load_fragA(const bf16_t* rowptr, int lane) {
  const int k0 = (lane >> 4) << 3;      // 0 or 8
  union { v16bf v; v8bf h[2]; } u;
  u.h[0] = *(const v8bf*)(rowptr + k0);
  u.h[1] = *(const v8bf*)(rowptr + k0 + 16);
  return u.v;
}
// B-matrix 32x16 bf16: lane holds column N = lane&15; K chunks of 8 at
// k0=(lane>>4)*8 and k0+16 (same K order as A).
__device__ __forceinline__ v16bf load_fragB(const bf16_t* colptr, int lane) {
  const int k0 = (lane >> 4) << 3;      // 0 or 8
  union { v16bf v; v8bf h[2]; } u;
  u.h[0] = *(const v8bf*)(colptr + k0);
  u.h[1] = *(const v8bf*)(colptr + k0 + 16);
  return u.v;
}
__device__ __forceinline__ v8f wmma_bf16(v16bf a, v16bf b, v8f c) {
  // (neg_a, A, neg_b, B, c_mod, C, reuse_a, reuse_b)
  return __builtin_amdgcn_wmma_f32_16x16x32_bf16(false, a, false, b, (short)0, c, false, false);
}

// ---------------- K0: transpose + convert x,c -> bf16 [b,t,c] ----------------
__global__ void k_transpose_convert(const float* __restrict__ x,
                                    const float* __restrict__ c,
                                    bf16_t* __restrict__ xT,
                                    bf16_t* __restrict__ cT) {
  __shared__ float tile[32][33];
  const int which = blockIdx.z >> 3;
  const int b     = blockIdx.z & 7;
  const float* src = which ? c : x;
  bf16_t*      dst = which ? cT : xT;
  const int t0 = blockIdx.x * 32, c0 = blockIdx.y * 32;
  const int tx = threadIdx.x, ty = threadIdx.y;
  const float* sb = src + (size_t)b * C_ * T_;
  #pragma unroll
  for (int r = 0; r < 4; ++r)
    tile[ty + 8*r][tx] = sb[(size_t)(c0 + ty + 8*r) * T_ + (t0 + tx)];
  __syncthreads();
  bf16_t* db = dst + (size_t)b * T_ * C_;
  #pragma unroll
  for (int r = 0; r < 4; ++r)
    db[(size_t)(t0 + ty + 8*r) * C_ + (c0 + tx)] = (bf16_t)tile[tx][ty + 8*r];
}

// ---------------- K1: weights f32 -> bf16 ----------------
__global__ void k_wconv(const float* __restrict__ wq, const float* __restrict__ wk,
                        const float* __restrict__ wv, const float* __restrict__ wo,
                        bf16_t* __restrict__ out) {
  const int m = blockIdx.y;
  const float* src = (m == 0) ? wq : (m == 1) ? wk : (m == 2) ? wv : wo;
  const size_t i = (size_t)blockIdx.x * blockDim.x + threadIdx.x;
  if (i < (size_t)C_ * C_) out[(size_t)m * C_ * C_ + i] = (bf16_t)src[i];
}

// ---------------- K2: QKV projection (WMMA bf16 GEMM) ----------------
// q -> qs[b,h,t,d] (scaled 1/8), k -> [b,h,t,d], v -> vT[b,h,d,t]
__global__ void k_qkv_proj(const bf16_t* __restrict__ wqkv,
                           const bf16_t* __restrict__ xT,
                           const bf16_t* __restrict__ cT,
                           const float* __restrict__ bq,
                           const float* __restrict__ bk,
                           const float* __restrict__ bv,
                           bf16_t* __restrict__ qs,
                           bf16_t* __restrict__ kk,
                           bf16_t* __restrict__ vT) {
  const int lane  = threadIdx.x;
  const int n0    = blockIdx.x * 64;     // t tile
  const int m0    = blockIdx.y * 16;     // o tile
  const int which = blockIdx.z % 3;      // 0=q 1=k 2=v
  const int b     = blockIdx.z / 3;
  const bf16_t* W  = wqkv + (size_t)which * C_ * C_;
  const bf16_t* Xb = ((which == 0) ? xT : cT) + (size_t)b * T_ * C_;

  v8f acc[4];
  #pragma unroll
  for (int nt = 0; nt < 4; ++nt) acc[nt] = (v8f){0,0,0,0,0,0,0,0};

  const int nlo  = lane & 15;
  const int arow = m0 + nlo;
  for (int k = 0; k < C_; k += 32) {
    v16bf a = load_fragA(W + (size_t)arow * C_ + k, lane);
    #pragma unroll
    for (int nt = 0; nt < 4; ++nt) {
      v16bf bb = load_fragB(Xb + (size_t)(n0 + nt * 16 + nlo) * C_ + k, lane);
      acc[nt] = wmma_bf16(a, bb, acc[nt]);
    }
  }

  const int half  = lane >> 4;
  const int obase = m0 + 8 * half;       // row M = obase + v
  const int h     = obase >> 6;
  const int d0    = obase & 63;
  if (which == 2) {
    for (int nt = 0; nt < 4; ++nt) {
      const int t = n0 + nt * 16 + nlo;
      #pragma unroll
      for (int v = 0; v < 8; ++v) {
        const float val = acc[nt][v] + bv[obase + v];
        vT[(((size_t)b * H_ + h) * 64 + (d0 + v)) * T_ + t] = (bf16_t)val;
      }
    }
  } else {
    const float* bias = (which == 0) ? bq : bk;
    const float scale = (which == 0) ? 0.125f : 1.0f;   // 1/sqrt(64)
    bf16_t* dst = (which == 0) ? qs : kk;
    float b8[8];
    #pragma unroll
    for (int v = 0; v < 8; ++v) b8[v] = bias[obase + v];
    for (int nt = 0; nt < 4; ++nt) {
      const int t = n0 + nt * 16 + nlo;
      v8bf pk;
      #pragma unroll
      for (int v = 0; v < 8; ++v) pk[v] = (bf16_t)((acc[nt][v] + b8[v]) * scale);
      *(v8bf*)(dst + (((size_t)b * H_ + h) * T_ + t) * 64 + d0) = pk;
    }
  }
}

// ---------------- K3: fused flash attention + banded rel-pos ----------------
// One wave = 16 query rows of one (b,h). Never materializes scores in HBM.
__global__ void k_attn(const bf16_t* __restrict__ qs,
                       const bf16_t* __restrict__ kk,
                       const bf16_t* __restrict__ vT,
                       const float* __restrict__ relk,
                       const float* __restrict__ relv,
                       bf16_t* __restrict__ attT) {
  __shared__ float  rband[16 * 9];
  __shared__ float  relv_s[9 * 64];
  __shared__ bf16_t pbuf[16 * 64];

  const int lane = threadIdx.x;
  const int t0   = blockIdx.x * 16;
  const int bh   = blockIdx.y;
  const int b    = bh / H_, h = bh % H_;

  const bf16_t* qh = qs + (size_t)bh * T_ * 64;
  const bf16_t* kh = kk + (size_t)bh * T_ * 64;
  const bf16_t* vh = vT + (size_t)bh * 64 * T_;

  for (int i = lane; i < 9 * 64; i += 32) relv_s[i] = relv[i];
  // per-row banded rel-key logits: rband[m][j] = qs[t0+m,:] . relk[j,:]
  for (int i = lane; i < 16 * 9; i += 32) {
    const int m = i / 9, j = i % 9;
    const bf16_t* qr = qh + (size_t)(t0 + m) * 64;
    const float*  rk = relk + j * 64;
    float s = 0.f;
    #pragma unroll 8
    for (int d = 0; d < 64; ++d) s += (float)qr[d] * rk[d];
    rband[i] = s;
  }

  const int nlo  = lane & 15;
  const int half = lane >> 4;

  v16bf qa[2];
  {
    const bf16_t* qrow = qh + (size_t)(t0 + nlo) * 64;
    qa[0] = load_fragA(qrow, lane);
    qa[1] = load_fragA(qrow + 32, lane);
  }
  __syncthreads();

  float mrow[8], lrow[8];
  v8f oacc[4];
  #pragma unroll
  for (int v = 0; v < 8; ++v) { mrow[v] = -3.0e38f; lrow[v] = 0.f; }
  #pragma unroll
  for (int dt = 0; dt < 4; ++dt) oacc[dt] = (v8f){0,0,0,0,0,0,0,0};

  for (int s0 = 0; s0 < T_; s0 += 64) {
    if (s0 + 64 < T_) {
      __builtin_prefetch(kh + (size_t)(s0 + 64 + nlo) * 64, 0, 0);
      __builtin_prefetch(vh + (size_t)nlo * T_ + s0 + 64, 0, 0);
    }
    // S = Q K^T (16 x 64), f32 accum
    v8f sacc[4];
    #pragma unroll
    for (int nt = 0; nt < 4; ++nt) {
      v8f a = (v8f){0,0,0,0,0,0,0,0};
      #pragma unroll
      for (int ks = 0; ks < 2; ++ks) {
        v16bf kb = load_fragB(kh + (size_t)(s0 + nt * 16 + nlo) * 64 + ks * 32, lane);
        a = wmma_bf16(qa[ks], kb, a);
      }
      sacc[nt] = a;
    }
    const bool band = (s0 <= t0 + 19) && (s0 + 63 >= t0 - 4);
    if (band) {
      #pragma unroll
      for (int nt = 0; nt < 4; ++nt) {
        const int s = s0 + nt * 16 + nlo;
        #pragma unroll
        for (int v = 0; v < 8; ++v) {
          const int M = v + 8 * half;
          const int r = s - (t0 + M) + 4;
          if (r >= 0 && r < 9) sacc[nt][v] += rband[M * 9 + r];
        }
      }
    }
    // online softmax (rows live in 16-lane halves; xor<16 stays in-half)
    #pragma unroll
    for (int v = 0; v < 8; ++v) {
      float mx = fmaxf(fmaxf(sacc[0][v], sacc[1][v]), fmaxf(sacc[2][v], sacc[3][v]));
      #pragma unroll
      for (int off = 1; off < 16; off <<= 1) mx = fmaxf(mx, __shfl_xor(mx, off, 32));
      const float mnew = fmaxf(mrow[v], mx);
      const float corr = __expf(mrow[v] - mnew);
      lrow[v] *= corr;
      #pragma unroll
      for (int dt = 0; dt < 4; ++dt) oacc[dt][v] *= corr;
      float rs = 0.f;
      #pragma unroll
      for (int nt = 0; nt < 4; ++nt) {
        const float p = __expf(sacc[nt][v] - mnew);
        sacc[nt][v] = p;
        rs += p;
      }
      #pragma unroll
      for (int off = 1; off < 16; off <<= 1) rs += __shfl_xor(rs, off, 32);
      lrow[v] += rs;
      mrow[v] = mnew;
    }
    // stage P (bf16) into LDS, re-load in A-layout
    #pragma unroll
    for (int nt = 0; nt < 4; ++nt)
      #pragma unroll
      for (int v = 0; v < 8; ++v)
        pbuf[(v + 8 * half) * 64 + nt * 16 + nlo] = (bf16_t)sacc[nt][v];
    __syncthreads();
    v16bf pa[2];
    pa[0] = load_fragA(pbuf + (size_t)nlo * 64, lane);
    pa[1] = load_fragA(pbuf + (size_t)nlo * 64 + 32, lane);
    // O += P V  (V stored transposed -> contiguous B-fragments)
    #pragma unroll
    for (int dt = 0; dt < 4; ++dt)
      #pragma unroll
      for (int ks = 0; ks < 2; ++ks) {
        v16bf vb = load_fragB(vh + (size_t)(dt * 16 + nlo) * T_ + s0 + ks * 32, lane);
        oacc[dt] = wmma_bf16(pa[ks], vb, oacc[dt]);
      }
    // banded rel-value contribution (unnormalized p, same scale as oacc)
    if (band) {
      #pragma unroll
      for (int v = 0; v < 8; ++v) {
        const int M = v + 8 * half;
        const int t = t0 + M;
        #pragma unroll
        for (int j = 0; j < 9; ++j) {
          const int s = t + j - 4;
          if (s >= s0 && s < s0 + 64) {
            const float pv = (float)pbuf[M * 64 + (s - s0)];
            #pragma unroll
            for (int dt = 0; dt < 4; ++dt)
              oacc[dt][v] += pv * relv_s[j * 64 + dt * 16 + nlo];
          }
        }
      }
    }
    __syncthreads();
  }
  // normalize, store attT[b, t, c]  (c = h*64 + d), bf16 for final GEMM
  bf16_t* ab = attT + (size_t)b * T_ * C_;
  #pragma unroll
  for (int v = 0; v < 8; ++v) {
    const float inv = 1.0f / lrow[v];
    const int t = t0 + v + 8 * half;
    #pragma unroll
    for (int dt = 0; dt < 4; ++dt) {
      const int cidx = h * 64 + dt * 16 + nlo;
      ab[(size_t)t * C_ + cidx] = (bf16_t)(oacc[dt][v] * inv);
    }
  }
}

// ---------------- K4: output projection (WMMA) -> f32 ----------------
__global__ void k_out_proj(const bf16_t* __restrict__ wo_bf,
                           const bf16_t* __restrict__ attT,
                           const float* __restrict__ bo,
                           float* __restrict__ y) {
  const int lane = threadIdx.x;
  const int n0   = blockIdx.x * 64;
  const int m0   = blockIdx.y * 16;
  const int b    = blockIdx.z;
  const bf16_t* Xb = attT + (size_t)b * T_ * C_;

  v8f acc[4];
  #pragma unroll
  for (int nt = 0; nt < 4; ++nt) acc[nt] = (v8f){0,0,0,0,0,0,0,0};
  const int nlo  = lane & 15;
  const int arow = m0 + nlo;
  for (int k = 0; k < C_; k += 32) {
    v16bf a = load_fragA(wo_bf + (size_t)arow * C_ + k, lane);
    #pragma unroll
    for (int nt = 0; nt < 4; ++nt) {
      v16bf bb = load_fragB(Xb + (size_t)(n0 + nt * 16 + nlo) * C_ + k, lane);
      acc[nt] = wmma_bf16(a, bb, acc[nt]);
    }
  }
  const int half  = lane >> 4;
  const int obase = m0 + 8 * half;
  float b8[8];
  #pragma unroll
  for (int v = 0; v < 8; ++v) b8[v] = bo[obase + v];
  for (int nt = 0; nt < 4; ++nt) {
    const int t = n0 + nt * 16 + nlo;
    #pragma unroll
    for (int v = 0; v < 8; ++v)
      y[((size_t)b * C_ + (obase + v)) * T_ + t] = acc[nt][v] + b8[v];
  }
}

// ---------------- host launch ----------------
extern "C" void kernel_launch(void* const* d_in, const int* in_sizes, int n_in,
                              void* d_out, int out_size, void* d_ws, size_t ws_size,
                              hipStream_t stream) {
  const float* x    = (const float*)d_in[0];
  const float* c    = (const float*)d_in[1];
  const float* wq   = (const float*)d_in[2];
  const float* bq   = (const float*)d_in[3];
  const float* wk   = (const float*)d_in[4];
  const float* bk   = (const float*)d_in[5];
  const float* wv   = (const float*)d_in[6];
  const float* bv   = (const float*)d_in[7];
  const float* wo   = (const float*)d_in[8];
  const float* bo   = (const float*)d_in[9];
  const float* relk = (const float*)d_in[10];  // [1,9,64]
  const float* relv = (const float*)d_in[11];  // [1,9,64]
  float* y = (float*)d_out;

  char* ws = (char*)d_ws;
  size_t off = 0;
  auto take = [&](size_t bytes) -> void* {
    void* p = ws + off;
    off += (bytes + 255) & ~(size_t)255;
    return p;
  };
  bf16_t* xT   = (bf16_t*)take((size_t)B_ * T_ * C_ * sizeof(bf16_t));
  bf16_t* cT   = (bf16_t*)take((size_t)B_ * T_ * C_ * sizeof(bf16_t));
  bf16_t* wbuf = (bf16_t*)take((size_t)4 * C_ * C_ * sizeof(bf16_t));   // q,k,v,o
  bf16_t* qsb  = (bf16_t*)take((size_t)B_ * H_ * T_ * KC_ * sizeof(bf16_t));
  bf16_t* kb   = (bf16_t*)take((size_t)B_ * H_ * T_ * KC_ * sizeof(bf16_t));
  bf16_t* vTb  = (bf16_t*)take((size_t)B_ * H_ * KC_ * T_ * sizeof(bf16_t));
  bf16_t* attT = (bf16_t*)take((size_t)B_ * T_ * C_ * sizeof(bf16_t));
  (void)ws_size; (void)in_sizes; (void)n_in; (void)out_size;

  k_transpose_convert<<<dim3(T_ / 32, C_ / 32, 2 * B_), dim3(32, 8), 0, stream>>>(x, c, xT, cT);
  k_wconv<<<dim3((C_ * C_ + 255) / 256, 4), 256, 0, stream>>>(wq, wk, wv, wo, wbuf);
  k_qkv_proj<<<dim3(T_ / 64, C_ / 16, 3 * B_), 32, 0, stream>>>(
      wbuf, xT, cT, bq, bk, bv, qsb, kb, vTb);
  k_attn<<<dim3(T_ / 16, B_ * H_), 32, 0, stream>>>(qsb, kb, vTb, relk, relv, attT);
  k_out_proj<<<dim3(T_ / 64, C_ / 16, B_), 32, 0, stream>>>(
      wbuf + (size_t)3 * C_ * C_, attT, bo, y);
}
